// lruz_45732811768498
// MI455X (gfx1250) — matrix operand order repaired and checked by reading, not compiled, not run on hardware
//
#include <hip/hip_runtime.h>

// ---------------------------------------------------------------------------
// LRU (complex diagonal SSM) for MI455X / gfx1250, wave32 + WMMA + TDM.
//   y = Re(pre_states @ C^T) + u @ D^T,  h_t = lam*h_{t-1} + (u_t @ B^T)
// B, C are REAL (Y_raw real in reference) => Bu and pre_re are real planes.
// GEMMs use split-f16 (Markidis 3-term) on v_wmma_f32_16x16x32_f16.
// f32->f16(hi,lo) splitting is done ONCE per element by cooperative phases
// that write LDS in native WMMA A-fragment layout; WMMA waves just ds_load.
//   k_gemm1_scan : u -> Bu (WMMA, staged in LDS) -> per-chunk end states
//   k_scanB      : sequential carry combine with lam^CHUNK
//   k_scan_gemm2 : Bu subtiles via tensor_load_to_lds (TDM, double buffered)
//                  -> scan to pre-fragments in LDS -> WMMA -> y
// ---------------------------------------------------------------------------

typedef __attribute__((ext_vector_type(16))) _Float16 v16h;
typedef __attribute__((ext_vector_type(8)))  float    v8f;
typedef __attribute__((ext_vector_type(4)))  unsigned int u32x4;
typedef __attribute__((ext_vector_type(8)))  unsigned int u32x8;

#define NST   256          // state dim N
#define NUD   128          // NU
#define NYD   128          // NY
#define BSZ   16           // batch
#define SEQL  4096         // sequence
#define MR    (BSZ*SEQL)   // 65536 GEMM rows
#define CHUNK 128          // scan chunk (rows per fused block)
#define NCH   (SEQL/CHUNK) // 32
#define SUB1  32           // producer subtile rows
#define NSUB1 (CHUNK/SUB1) // 4
#define SUB2  16           // consumer subtile rows
#define NSUB2 (CHUNK/SUB2) // 8

// ---------------- WMMA helpers -------------------------------------------

__device__ inline v8f wmma3(v16h ah, v16h al, v16h bh, v16h bl, v8f acc) {
  acc = __builtin_amdgcn_wmma_f32_16x16x32_f16(false, ah, false, bh, (short)0, acc, false, false);
  acc = __builtin_amdgcn_wmma_f32_16x16x32_f16(false, al, false, bh, (short)0, acc, false, false);
  acc = __builtin_amdgcn_wmma_f32_16x16x32_f16(false, ah, false, bl, (short)0, acc, false, false);
  return acc;
}

// A-fragment element address for 16x32 f16 A tile (ISA 05_wmma):
//   lane = M + 16*((Kl>>3)&1),  elem = (Kl&7) + 8*((Kl>>4)&1),  Kl = K%32
__device__ inline int afrag_idx(int M, int Kl) {
  int lane = M + 16 * ((Kl >> 3) & 1);
  int elem = (Kl & 7) + 8 * ((Kl >> 4) & 1);
  return lane * 16 + elem;
}

// split one f32 into hi/lo f16 and store into fragment-layout LDS arrays
__device__ inline void split_store(_Float16* fh, _Float16* fl, int idx, float x) {
  _Float16 h = (_Float16)x;
  fh[idx] = h;
  fl[idx] = (_Float16)(x - (float)h);
}

// ---------------- TDM: 1-D tile load global->LDS -------------------------
// D# group0: count=1 | lds_addr | global_addr | type=2 ("image")
// D# group1: data_size=4B, tensor_dim0=tile_dim0=nfloats, tensor_dim1=tile_dim1=1
__device__ inline void tdm_load_tile(const float* gsrc, float* lds_dst, int nfloats) {
  unsigned long long ga = (unsigned long long)(uintptr_t)gsrc;
  unsigned int la = (unsigned int)(uintptr_t)lds_dst;   // low 32 bits = LDS offset
  u32x4 g0; u32x8 g1;
  g0[0] = 1u;                                            // count=1, user mode
  g0[1] = la;                                            // lds_addr
  g0[2] = (unsigned int)ga;                              // global_addr[31:0]
  g0[3] = (unsigned int)(ga >> 32) | (2u << 30);         // global_addr[56:32] | type=2
  unsigned int nf = (unsigned int)nfloats;
  g1[0] = (2u << 16);                                    // data_size = 4 bytes
  g1[1] = (nf & 0xFFFFu) << 16;                          // tensor_dim0[15:0]
  g1[2] = ((nf >> 16) & 0xFFFFu) | (1u << 16);           // tensor_dim0[31:16] | tensor_dim1=1
  g1[3] = (nf & 0xFFFFu) << 16;                          // tile_dim0
  g1[4] = 1u;                                            // tile_dim1=1 (tile_dim2=0)
  g1[5] = 0u; g1[6] = 0u; g1[7] = 0u;                    // strides unused (1 row)
  asm volatile("tensor_load_to_lds %0, %1" :: "s"(g0), "s"(g1) : "memory");
}

// ---------------- setup kernel (1 block) ---------------------------------

__device__ inline float blockSum(float v, float* red) {
  int tid = threadIdx.x;
  red[tid] = v; __syncthreads();
  for (int s = 128; s > 0; s >>= 1) { if (tid < s) red[tid] += red[tid + s]; __syncthreads(); }
  float r = red[0]; __syncthreads();
  return r;
}

__global__ __launch_bounds__(256)
void k_setup(const float* __restrict__ nu_log, const float* __restrict__ theta_log,
             const float* __restrict__ gamma_raw, const float* __restrict__ X2b,
             const float* __restrict__ Dp,
             float* __restrict__ lamRe, float* __restrict__ lamIm,
             float* __restrict__ lamPR, float* __restrict__ lamPI,
             float* __restrict__ Sg,
             _Float16* __restrict__ bfh, _Float16* __restrict__ bfl,
             _Float16* __restrict__ wfh, _Float16* __restrict__ wfl) {
  __shared__ float sh_lr[256], sh_li[256], sh_s[256];
  __shared__ float sh_vR[256], sh_vI[256];
  __shared__ float sh_aR[512], sh_aI[512];
  __shared__ float sh_bR[512], sh_bI[512];
  __shared__ float sh_x1R[128], sh_x1I[128], sh_x2R[128], sh_x2I[128];
  __shared__ float sh_px[128], sh_py[128];
  __shared__ float sh_red[256];
  __shared__ float sc[8];   // 0:gamma 1:d_norm 2:nrm 3:sD 4:sY
  int tid = threadIdx.x;

  // ---- lam, s = sqrt(1-|lam|^2), lam^CHUNK -------------------------------
  {
    float nu  = expf(nu_log[tid]);
    float th  = expf(theta_log[tid]);
    float mag = expf(-nu);
    float lr = mag * cosf(th), li = mag * sinf(th);
    sh_lr[tid] = lr; sh_li[tid] = li;
    sh_s[tid]  = sqrtf(fmaxf(1.f - (lr*lr + li*li), 1e-12f));
    lamRe[tid] = lr; lamIm[tid] = li;
    float zr = lr, zi = li;                 // lam^128 = 7 squarings
    for (int q = 0; q < 7; ++q) { float nr = zr*zr - zi*zi; zi = 2.f*zr*zi; zr = nr; }
    lamPR[tid] = zr; lamPI[tid] = zi;
  }
  if (tid == 0) {
    float g = gamma_raw[0];
    sc[0] = ((g > 20.f) ? g : log1pf(expf(g))) + 1e-6f;   // softplus + eps
  }
  __syncthreads();

  // ---- d_norm = ||Dp||_2 via power iteration on Dp^T Dp ------------------
  if (tid < 128) sh_px[tid] = 1.f + 0.001f * tid;
  __syncthreads();
  for (int it = 0; it < 40; ++it) {
    float yv = 0.f;
    if (tid < 128) { for (int k = 0; k < 128; ++k) yv += Dp[tid*128 + k] * sh_px[k]; sh_py[tid] = yv; }
    __syncthreads();
    float xv = 0.f;
    if (tid < 128) { for (int j = 0; j < 128; ++j) xv += Dp[j*128 + tid] * sh_py[j]; }
    float nn = blockSum((tid < 128) ? xv*xv : 0.f, sh_red);
    float inv = rsqrtf(fmaxf(nn, 1e-30f));
    if (tid < 128) sh_px[tid] = xv * inv;
    __syncthreads();
  }
  {
    float yv = 0.f;
    if (tid < 128) { for (int k = 0; k < 128; ++k) yv += Dp[tid*128 + k] * sh_px[k]; }
    float nn = blockSum((tid < 128) ? yv*yv : 0.f, sh_red);
    if (tid == 0) {
      sc[1] = sqrtf(fmaxf(nn, 0.f));
      sc[3] = fminf(1.f, sc[0] * 0.95f / fmaxf(sc[1], 1e-12f));   // sD
    }
  }
  __syncthreads();

  const float gam = sc[0], sD = sc[3];
  const float sg  = sqrtf(gam);

  // ---- S = gam*I - (sD^2/gam) Dp Dp^T  (in ws), then Cholesky ------------
  {
    float c = sD * sD / gam;
    for (int idx = tid; idx < 128*128; idx += 256) {
      int i = idx >> 7, j = idx & 127;
      float acc = 0.f;
      for (int k = 0; k < 128; ++k) acc += Dp[i*128 + k] * Dp[j*128 + k];
      Sg[idx] = ((i == j) ? gam : 0.f) - c * acc;
    }
  }
  __syncthreads();
  for (int k = 0; k < 128; ++k) {
    if (tid == 0) Sg[k*128 + k] = sqrtf(fmaxf(Sg[k*128 + k], 1e-20f));
    __syncthreads();
    float dk = Sg[k*128 + k];
    for (int i = k + 1 + tid; i < 128; i += 256) Sg[i*128 + k] /= dk;
    __syncthreads();
    for (int i = k + 1; i < 128; ++i) {
      float lik = Sg[i*128 + k];
      for (int j = k + 1 + tid; j <= i; j += 256) Sg[i*128 + j] -= lik * Sg[j*128 + k];
    }
    __syncthreads();
  }

  // ---- nrm = ||Lw^{-1} Y Lz^{-H}||_2 via power iteration on A^H A --------
  sh_vR[tid] = 1.f + 0.0013f * tid;
  sh_vI[tid] = 0.5f - 0.0007f * tid;
  __syncthreads();
  {
    float nn = blockSum(sh_vR[tid]*sh_vR[tid] + sh_vI[tid]*sh_vI[tid], sh_red);
    float inv = rsqrtf(fmaxf(nn, 1e-30f));
    sh_vR[tid] *= inv; sh_vI[tid] *= inv;
  }
  __syncthreads();

  for (int it = 0; it < 32; ++it) {
    if (tid < 128) { sh_x2R[tid] = sh_vR[128 + tid]; sh_x2I[tid] = sh_vI[128 + tid]; }
    __syncthreads();
    for (int i = 127; i >= 0; --i) {
      if (tid == 0) { float d = Sg[i*128 + i]; sh_x2R[i] /= d; sh_x2I[i] /= d; }
      __syncthreads();
      float xr = sh_x2R[i], xi = sh_x2I[i];
      int j = tid & 127;
      if (j < i) { float l = Sg[i*128 + j]; if (tid < 128) sh_x2R[j] -= l*xr; else sh_x2I[j] -= l*xi; }
      __syncthreads();
    }
    if (tid < 128) {
      float dr = 0.f, di = 0.f;
      for (int j = 0; j < 128; ++j) { float d = Dp[j*128 + tid]; dr += d*sh_x2R[j]; di += d*sh_x2I[j]; }
      sh_x1R[tid] = (sh_vR[tid] - sD*dr/sg) / sg;
      sh_x1I[tid] = (sh_vI[tid] - sD*di/sg) / sg;
    }
    __syncthreads();
    {
      float wr = 0.f, wi = 0.f;
      for (int k = 0; k < 128; ++k) { float bv = X2b[tid*256 + k]; wr += bv*sh_x1R[k]; wi += bv*sh_x1I[k]; }
      sh_aR[tid] = wr; sh_aI[tid] = wi;
      wr = 0.f; wi = 0.f;
      for (int k = 0; k < 128; ++k) { float cv = X2b[(256 + tid)*256 + 128 + k]; wr += cv*sh_x2R[k]; wi += cv*sh_x2I[k]; }
      sh_aR[256 + tid] = wr; sh_aI[256 + tid] = wi;
    }
    __syncthreads();
    {
      float lr = sh_lr[tid], li = sh_li[tid], sv = sh_s[tid];
      float w1r = sh_aR[tid], w1i = sh_aI[tid];
      float w2r = sh_aR[256 + tid], w2i = sh_aI[256 + tid];
      float z2r = (w2r - (lr*w1r + li*w1i)) / sv;          // conj(lam)*w1
      float z2i = (w2i - (lr*w1i - li*w1r)) / sv;
      float t2r = z2r / sv, t2i = z2i / sv;
      float t1r = w1r - (lr*t2r - li*t2i);                 // lam*t2
      float t1i = w1i - (lr*t2i + li*t2r);
      sh_bR[tid] = t1r; sh_bI[tid] = t1i;
      sh_bR[256 + tid] = t2r; sh_bI[256 + tid] = t2i;
    }
    __syncthreads();
    {
      int k = tid & 127;
      float rr = 0.f, ri = 0.f;
      if (tid < 128) {
        for (int n = 0; n < 256; ++n) { float bv = X2b[n*256 + k]; rr += bv*sh_bR[n]; ri += bv*sh_bI[n]; }
        sh_x1R[k] = rr / sg; sh_x1I[k] = ri / sg;
      } else {
        for (int n = 0; n < 256; ++n) { float cv = X2b[(256 + n)*256 + 128 + k]; rr += cv*sh_bR[256 + n]; ri += cv*sh_bI[256 + n]; }
        sh_x2R[k] = rr; sh_x2I[k] = ri;
      }
    }
    __syncthreads();
    if (tid < 128) {
      float dr = 0.f, di = 0.f;
      for (int k = 0; k < 128; ++k) { float d = Dp[tid*128 + k]; dr += d*sh_x1R[k]; di += d*sh_x1I[k]; }
      sh_x2R[tid] -= sD*dr/sg; sh_x2I[tid] -= sD*di/sg;
    }
    __syncthreads();
    for (int i = 0; i < 128; ++i) {
      if (tid == 0) { float d = Sg[i*128 + i]; sh_x2R[i] /= d; sh_x2I[i] /= d; }
      __syncthreads();
      float xr = sh_x2R[i], xi = sh_x2I[i];
      int j = tid & 127;
      if (j > i) { float l = Sg[j*128 + i]; if (tid < 128) sh_x2R[j] -= l*xr; else sh_x2I[j] -= l*xi; }
      __syncthreads();
    }
    float val = 0.f;
    if (tid < 128) val = sh_x1R[tid]*sh_x1R[tid] + sh_x1I[tid]*sh_x1I[tid]
                       + sh_x2R[tid]*sh_x2R[tid] + sh_x2I[tid]*sh_x2I[tid];
    float nn = blockSum(val, sh_red);
    if (tid == 0) sc[2] = sqrtf(sqrtf(fmaxf(nn, 1e-30f)));   // sigma = ||A^H A v||^{1/2}
    float inv = rsqrtf(fmaxf(nn, 1e-30f));
    if (tid < 128) { sh_vR[tid] = sh_x1R[tid]*inv; sh_vI[tid] = sh_x1I[tid]*inv; }
    else           { sh_vR[tid] = sh_x2R[tid-128]*inv; sh_vI[tid] = sh_x2I[tid-128]*inv; }
    __syncthreads();
  }
  if (tid == 0) sc[4] = fminf(1.f, 0.9f / fmaxf(sc[2], 1e-12f));   // sY
  __syncthreads();
  const float sY = sc[4];

  // ---- pack GEMM1 weight fragments: W1[K][n] = sY * X2b[n*256+K], K<128 --
  for (int idx = tid; idx < 4*16*512; idx += 256) {
    int tile = idx >> 9, lane = (idx >> 4) & 31, e = idx & 15;
    int kt = tile >> 4, ntg = tile & 15;
    int K = kt*32 + (lane >> 4)*8 + (e & 7) + (e >> 3)*16;
    int n = ntg*16 + (lane & 15);
    float val = sY * X2b[n*256 + K];
    _Float16 h = (_Float16)val;
    bfh[idx] = h; bfl[idx] = (_Float16)(val - (float)h);
  }
  // ---- pack GEMM2 weights: Wcat[K][j] = C[j][K] (K<256) | D[j][K-256] ----
  for (int idx = tid; idx < 12*8*512; idx += 256) {
    int tile = idx >> 9, lane = (idx >> 4) & 31, e = idx & 15;
    int kt = tile >> 3, ntg = tile & 7;
    int K = kt*32 + (lane >> 4)*8 + (e & 7) + (e >> 3)*16;
    int j = ntg*16 + (lane & 15);
    float val = (K < 256) ? sY * X2b[(256 + K)*256 + 128 + j]
                          : sD * Dp[j*128 + (K - 256)];
    _Float16 h = (_Float16)val;
    wfh[idx] = h; wfl[idx] = (_Float16)(val - (float)h);
  }
}

// ---------------- fused GEMM1 + local scan -------------------------------
// One block per (batch, chunk).  Cooperative phase splits the u subtile into
// hi/lo f16 A-fragments in LDS (coalesced reads, one conversion per element);
// WMMA waves ds_load fragments, write Bu to LDS + plane; in-block scan.

__global__ __launch_bounds__(256)
void k_gemm1_scan(const float* __restrict__ u,
                  const _Float16* __restrict__ bfh, const _Float16* __restrict__ bfl,
                  const float* __restrict__ lamRe, const float* __restrict__ lamIm,
                  float* __restrict__ plane,
                  float* __restrict__ lendR, float* __restrict__ lendI) {
  __shared__ float   buf[SUB1 * NST];       // 32 KB  Bu subtile
  __shared__ _Float16 ufh[8 * 512];         // 8 KB   u A-fragments hi (2 Mtiles x 4 kt)
  __shared__ _Float16 ufl[8 * 512];         // 8 KB   u A-fragments lo
  int tid = threadIdx.x, lane = tid & 31, wid = tid >> 5;
  int b = blockIdx.x / NCH, c = blockIdx.x % NCH;
  int chunkRow = b * SEQL + c * CHUNK;
  int n = tid;
  float lr = lamRe[n], li = lamIm[n];
  float hr = 0.f, hi = 0.f;
  int mrowL = (wid & 1) * 16;
  int nt0   = (wid >> 1) * 2;

  for (int s = 0; s < NSUB1; ++s) {
    int rowG0 = chunkRow + s * SUB1;
    // ---- cooperative u split: 32 rows x 128 cols, 16 floats per thread ----
    {
      int row = tid >> 3;               // 0..31
      int k0  = (tid & 7) * 16;         // 16 consecutive cols
      const float* up = u + (size_t)(rowG0 + row) * NUD + k0;
      int mt = row >> 4, M = row & 15;
#pragma unroll
      for (int i = 0; i < 16; ++i) {
        int K = k0 + i;
        int fidx = (mt * 4 + (K >> 5)) * 512 + afrag_idx(M, K & 31);
        split_store(ufh, ufl, fidx, up[i]);
      }
    }
    __syncthreads();
    // ---- WMMA: wave w -> Mtile (w&1), n-tiles nt0..nt0+1 ------------------
    {
      v16h ah[4], al[4];
#pragma unroll
      for (int kt = 0; kt < 4; ++kt) {
        ah[kt] = ((const v16h*)ufh)[((wid & 1) * 4 + kt) * 32 + lane];
        al[kt] = ((const v16h*)ufl)[((wid & 1) * 4 + kt) * 32 + lane];
      }
#pragma unroll
      for (int nt = 0; nt < 2; ++nt) {
        int ntg = nt0 + nt;
        v8f acc = {};
#pragma unroll
        for (int kt = 0; kt < 4; ++kt) {
          int tile = kt * 16 + ntg;
          v16h bh = ((const v16h*)bfh)[tile * 32 + lane];
          v16h bl = ((const v16h*)bfl)[tile * 32 + lane];
          acc = wmma3(ah[kt], al[kt], bh, bl, acc);
        }
        int col = ntg * 16 + (lane & 15);
        int rbL = mrowL + (lane >> 4) * 8;
#pragma unroll
        for (int r = 0; r < 8; ++r) {
          plane[(size_t)(rowG0 + rbL + r) * NST + col] = acc[r];
          buf[(rbL + r) * NST + col] = acc[r];
        }
      }
    }
    __syncthreads();
    // ---- in-block local scan over this subtile (thread n = state n) ------
#pragma unroll 4
    for (int t = 0; t < SUB1; ++t) {
      float bu = buf[t * NST + n];
      float nr = lr * hr - li * hi + bu;
      float ni = lr * hi + li * hr;
      hr = nr; hi = ni;
    }
    __syncthreads();
  }
  lendR[blockIdx.x * NST + n] = hr;
  lendI[blockIdx.x * NST + n] = hi;
}

// ---------------- scan phase B: sequential carry combine -----------------

__global__ __launch_bounds__(256)
void k_scanB(const float* __restrict__ lamPR, const float* __restrict__ lamPI,
             const float* __restrict__ lendR, const float* __restrict__ lendI,
             float* __restrict__ cinR, float* __restrict__ cinI) {
  int b = blockIdx.x, n = threadIdx.x;
  float Lr = lamPR[n], Li = lamPI[n];
  float cr = 0.f, ci = 0.f;
  for (int c = 0; c < NCH; ++c) {
    int idx = (b * NCH + c) * NST + n;
    cinR[idx] = cr; cinI[idx] = ci;
    float er = lendR[idx], ei = lendI[idx];
    float nr = Lr * cr - Li * ci + er;
    float ni = Lr * ci + Li * cr + ei;
    cr = nr; ci = ni;
  }
}

// ---------------- fused scan + GEMM2 -------------------------------------
// One block per (batch, chunk).  TDM-streams 16-row Bu subtiles into a
// double-buffered LDS tile; the scan threads convert pre_states directly
// into hi/lo f16 A-fragments in LDS (one conversion per element); a
// cooperative phase does the same for the u subtile; WMMA waves ds_load
// fragments and write y.

__global__ __launch_bounds__(256)
void k_scan_gemm2(const float* __restrict__ plane, const float* __restrict__ u,
                  const _Float16* __restrict__ wfh, const _Float16* __restrict__ wfl,
                  const float* __restrict__ lamRe, const float* __restrict__ lamIm,
                  const float* __restrict__ cinR, const float* __restrict__ cinI,
                  float* __restrict__ y) {
  __shared__ float   buf[2][SUB2 * NST];    // 32 KB  Bu subtiles (double buffered)
  __shared__ _Float16 pfh[8 * 512];         // 8 KB   pre A-fragments hi (kt 0..7)
  __shared__ _Float16 pfl[8 * 512];         // 8 KB   pre A-fragments lo
  __shared__ _Float16 ufh[4 * 512];         // 4 KB   u A-fragments hi (kt 8..11)
  __shared__ _Float16 ufl[4 * 512];         // 4 KB   u A-fragments lo
  int tid = threadIdx.x, lane = tid & 31, wid = tid >> 5;
  int b = blockIdx.x / NCH, c = blockIdx.x % NCH;
  int chunkRow = b * SEQL + c * CHUNK;

  int n = tid;
  float lr = lamRe[n], li = lamIm[n];
  int cidx = blockIdx.x * NST + n;
  float hr = cinR[cidx], hi = cinI[cidx];
  int pktbase = (n >> 5) * 512;             // this thread's pre fragment
  int lhalf   = 16 * ((n >> 3) & 1);
  int pelem   = (n & 7) + 8 * ((n >> 4) & 1);

  // prologue: TDM-load subtile 0 (wave 0 issues; TDM ignores EXEC)
  if (wid == 0) tdm_load_tile(plane + (size_t)chunkRow * NST, &buf[0][0], SUB2 * NST);

  for (int s = 0; s < NSUB2; ++s) {
    if (wid == 0) {
      if (s < NSUB2 - 1) {
        tdm_load_tile(plane + (size_t)(chunkRow + (s + 1) * SUB2) * NST,
                      &buf[(s + 1) & 1][0], SUB2 * NST);
        __builtin_amdgcn_s_wait_tensorcnt(1);   // subtile s complete
      } else {
        __builtin_amdgcn_s_wait_tensorcnt(0);   // last subtile complete
      }
    }
    __syncthreads();                            // publish TDM data; frags reusable
    const float* bp = &buf[s & 1][0];
    int rowG = chunkRow + s * SUB2;

    // ---- scan subtile; emit pre_re as split-f16 A-fragments ---------------
#pragma unroll 4
    for (int t = 0; t < SUB2; ++t) {
      float bu = bp[t * NST + n];
      split_store(pfh, pfl, pktbase + (t + lhalf) * 16 + pelem, hr);  // pre = h_{t-1}
      float nr = lr * hr - li * hi + bu;
      float ni = lr * hi + li * hr;
      hr = nr; hi = ni;
    }
    // ---- cooperative u split: 16 rows x 128 cols, 8 floats per thread -----
    {
      int row = tid >> 4;               // 0..15
      int k0  = (tid & 15) * 8;         // 8 consecutive cols
      const float* up = u + (size_t)(rowG + row) * NUD + k0;
#pragma unroll
      for (int i = 0; i < 8; ++i) {
        int K = k0 + i;
        int fidx = (K >> 5) * 512 + afrag_idx(row, K & 31);
        split_store(ufh, ufl, fidx, up[i]);
      }
    }
    __syncthreads();

    // ---- WMMA on this 16-row subtile: wave `wid` owns output n-tile `wid` -
    v8f acc = {};
#pragma unroll
    for (int kt = 0; kt < 12; ++kt) {
      v16h ah, al;
      if (kt < 8) { ah = ((const v16h*)pfh)[kt * 32 + lane];
                    al = ((const v16h*)pfl)[kt * 32 + lane]; }
      else        { ah = ((const v16h*)ufh)[(kt - 8) * 32 + lane];
                    al = ((const v16h*)ufl)[(kt - 8) * 32 + lane]; }
      int tile = kt * 8 + wid;
      v16h bh = ((const v16h*)wfh)[tile * 32 + lane];
      v16h bl = ((const v16h*)wfl)[tile * 32 + lane];
      acc = wmma3(ah, al, bh, bl, acc);
    }
    int col = wid * 16 + (lane & 15);
    int rb  = rowG + (lane >> 4) * 8;
#pragma unroll
    for (int r = 0; r < 8; ++r) y[(size_t)(rb + r) * NYD + col] = acc[r];
  }
}

// ---------------- launcher ----------------------------------------------

extern "C" void kernel_launch(void* const* d_in, const int* in_sizes, int n_in,
                              void* d_out, int out_size, void* d_ws, size_t ws_size,
                              hipStream_t stream) {
  const float* u         = (const float*)d_in[0];
  const float* nu_log    = (const float*)d_in[1];
  const float* theta_log = (const float*)d_in[2];
  const float* gamma_raw = (const float*)d_in[3];
  const float* X2b       = (const float*)d_in[4];
  const float* Dp        = (const float*)d_in[5];
  float* y = (float*)d_out;

  // workspace carve-up (plane = 64 MB, params ~2.5 MB)
  float* plane = (float*)d_ws;
  float* lamRe = plane + (size_t)MR * NST;
  float* lamIm = lamRe + 256;
  float* lamPR = lamRe + 512;
  float* lamPI = lamRe + 768;
  float* lendR = lamRe + 1024;
  float* lendI = lendR + BSZ * NCH * NST;
  float* cinR  = lendI + BSZ * NCH * NST;
  float* cinI  = cinR  + BSZ * NCH * NST;
  float* Sg    = cinI  + BSZ * NCH * NST;
  _Float16* bfh = (_Float16*)(Sg + 128 * 128);
  _Float16* bfl = bfh + 4 * 16 * 512;
  _Float16* wfh = bfl + 4 * 16 * 512;
  _Float16* wfl = wfh + 12 * 8 * 512;

  k_setup<<<1, 256, 0, stream>>>(nu_log, theta_log, gamma_raw, X2b, Dp,
                                 lamRe, lamIm, lamPR, lamPI, Sg,
                                 bfh, bfl, wfh, wfl);
  k_gemm1_scan<<<BSZ * NCH, 256, 0, stream>>>(u, bfh, bfl, lamRe, lamIm,
                                              plane, lendR, lendI);
  k_scanB<<<BSZ, 256, 0, stream>>>(lamPR, lamPI, lendR, lendI, cinR, cinI);
  k_scan_gemm2<<<BSZ * NCH, 256, 0, stream>>>(plane, u, wfh, wfl,
                                              lamRe, lamIm, cinR, cinI, y);
}
